// Attention_24361054502915
// MI455X (gfx1250) — compile-verified, matrix operand order checked by reading, not run on hardware
//
#include <hip/hip_runtime.h>
#include <math.h>

// ---------------------------------------------------------------------------
// Multi-head self-attention, f16 WMMA (f32 accumulate) pipeline for gfx1250.
// B=4, N=2048, DIM=768, HEADS=12, DHEAD=64. scale 1/8 folded into w_q (f16).
// All GEMM tile staging uses gfx1250 async memory->LDS DMA (ASYNCcnt).
// ---------------------------------------------------------------------------

typedef __attribute__((ext_vector_type(4)))  float    f32x4;
typedef __attribute__((ext_vector_type(4)))  int      i32x4;
typedef __attribute__((ext_vector_type(8)))  _Float16 h8;
typedef __attribute__((ext_vector_type(16))) _Float16 v16h;
typedef __attribute__((ext_vector_type(8)))  float    v8f;

#define DIMF   768
#define NSEQ   2048
#define BATCH  4
#define NHEAD  12
#define DHEAD  64
#define MROWS  (BATCH * NSEQ)   /* 8192 token rows */

#if __has_builtin(__builtin_amdgcn_global_load_async_to_lds_b128) && \
    __has_builtin(__builtin_amdgcn_s_wait_asynccnt)
#define HAVE_ASYNC_LDS 1
#endif

// 16-byte memory->LDS tile copy: async (no VGPR bounce, ASYNCcnt) when the
// gfx1250 async builtins are available, else plain load/store.
// Builtin signature (from hipcc diagnostic): (int4 AS1*, int4 AS3*, Imm, Imm).
__device__ __forceinline__ void copy16_to_lds(_Float16* lds, const _Float16* g) {
#ifdef HAVE_ASYNC_LDS
    __builtin_amdgcn_global_load_async_to_lds_b128(
        (__attribute__((address_space(1))) i32x4*)g,
        (__attribute__((address_space(3))) i32x4*)lds, 0, 0);
#else
    *(h8*)lds = *(const h8*)g;
#endif
}
__device__ __forceinline__ void async_fence() {
#ifdef HAVE_ASYNC_LDS
    __builtin_amdgcn_s_wait_asynccnt(0);
#endif
}

// Load one 16x32 f16 WMMA operand fragment (A layout; B uses same shape with
// lane = N column) from a K-contiguous row-major tile of given stride (halves).
// Lane l (l<16): row=base+l, halves = K[kb..kb+7] ++ K[kb+16..kb+23], kb=0.
// Lane l>=16: same row, kb=8 -> K[8..15] ++ K[24..31]. Matches ISA 7.12.2.
__device__ __forceinline__ v16h load_frag(const _Float16* p, int rowbase, int stride) {
    const int lane = threadIdx.x & 31;
    const int r    = rowbase + (lane & 15);
    const int kb   = (lane >> 4) * 8;
    const _Float16* q = p + (size_t)r * stride + kb;
    h8 lo = *(const h8*)(q);        // 16B aligned: ds_load_b128 / global_load_b128
    h8 hi = *(const h8*)(q + 16);
    return __builtin_shufflevector(lo, hi, 0,1,2,3,4,5,6,7,8,9,10,11,12,13,14,15);
}

__device__ __forceinline__ v8f wmma(v16h a, v16h b, v8f c) {
    return __builtin_amdgcn_wmma_f32_16x16x32_f16(false, a, false, b, (short)0, c,
                                                  false, false);
}

// ---------------------------------------------------------------------------
// Kernel 1a: convert+transpose weights to f16, N-major ([out][in], in-contig).
// z=0..3 -> w_q (scaled 1/8), w_k, w_v, w_o.
// ---------------------------------------------------------------------------
__global__ __launch_bounds__(256) void convert_weights(
        const float* __restrict__ wq, const float* __restrict__ wk,
        const float* __restrict__ wv, const float* __restrict__ wo,
        _Float16* __restrict__ wT) {
    const int z = blockIdx.y;
    const float* src   = (z == 0) ? wq : (z == 1) ? wk : (z == 2) ? wv : wo;
    const float  scale = (z == 0) ? 0.125f : 1.0f;   // fold 1/sqrt(64) into Q
    const int idx = blockIdx.x * 256 + threadIdx.x;  // 0 .. 768*768-1
    const int n = idx / DIMF;
    const int k = idx - n * DIMF;
    wT[(size_t)z * DIMF * DIMF + (size_t)n * DIMF + k] =
        (_Float16)(src[(size_t)k * DIMF + n] * scale);
}

// ---------------------------------------------------------------------------
// Kernel 1b: convert activations x (f32) -> x16 (f16), streaming pass.
// Removes f32->f16 conversion from the QKV GEMM hot loop (pure async staging).
// ---------------------------------------------------------------------------
__global__ __launch_bounds__(256) void convert_x(
        const float* __restrict__ x, _Float16* __restrict__ x16) {
    const size_t i = ((size_t)blockIdx.x * 256 + threadIdx.x) * 8;
    f32x4 a = *(const f32x4*)(x + i);
    f32x4 b = *(const f32x4*)(x + i + 4);
    h8 o;
#pragma unroll
    for (int j = 0; j < 4; ++j) { o[j] = (_Float16)a[j]; o[4 + j] = (_Float16)b[j]; }
    *(h8*)(x16 + i) = o;
}

// ---------------------------------------------------------------------------
// Kernel 2: QKV projection GEMM.  C[8192x768] = x16 @ W.
// Workgroup tile 256x64, wave tile 64x32 (4x2 WMMA frags), K-step 32.
// z=0 -> Q [b,h,n,d], z=1 -> K [b,h,n,d], z=2 -> V transposed [b,h,d,n].
// ---------------------------------------------------------------------------
__global__ __launch_bounds__(256) void qkv_gemm(
        const _Float16* __restrict__ x16, const _Float16* __restrict__ wT,
        _Float16* __restrict__ Qo, _Float16* __restrict__ Ko,
        _Float16* __restrict__ Vt) {
    __shared__ _Float16 As[256 * 32];   // [m][k], k contiguous (16 KB)
    __shared__ _Float16 Bs[64 * 32];    // [n][k], k contiguous (4 KB)

    const int z    = blockIdx.z;
    const int n0   = blockIdx.x * 64;
    const int m0   = blockIdx.y * 256;
    const int tid  = threadIdx.x;
    const int wave = tid >> 5, lane = tid & 31;
    const int mw   = (wave >> 1) * 64;      // 4 waves along M
    const int nw   = (wave & 1) * 32;       // 2 waves along N
    const _Float16* w = wT + (size_t)z * DIMF * DIMF;

    v8f c[4][2] = {};
    for (int k0 = 0; k0 < DIMF; k0 += 32) {
        __syncthreads();
        // stage A: 256x32 halves, 4 async b128 per thread (no VGPR bounce)
#pragma unroll
        for (int i = 0; i < 4; ++i) {
            int idx = tid + i * 256;
            int row = idx >> 2, seg = (idx & 3) * 8;
            copy16_to_lds(&As[row * 32 + seg],
                          x16 + (size_t)(m0 + row) * DIMF + k0 + seg);
        }
        // stage B: 64x32 halves, 1 async b128 per thread
        {
            int row = tid >> 2, seg = (tid & 3) * 8;
            copy16_to_lds(&Bs[row * 32 + seg],
                          w + (size_t)(n0 + row) * DIMF + k0 + seg);
        }
        if (k0 + 32 < DIMF) {   // prefetch next K-slab (global_prefetch_b8)
            __builtin_prefetch(x16 + (size_t)(m0 + (tid >> 2)) * DIMF + k0 + 32 + (tid & 3) * 8, 0, 3);
            __builtin_prefetch(w + (size_t)(n0 + (tid >> 2)) * DIMF + k0 + 32 + (tid & 3) * 8, 0, 3);
        }
        async_fence();
        __syncthreads();
        v16h b0 = load_frag(Bs, nw +  0, 32);
        v16h b1 = load_frag(Bs, nw + 16, 32);
#pragma unroll
        for (int mi = 0; mi < 4; ++mi) {
            v16h a = load_frag(As, mw + mi * 16, 32);
            c[mi][0] = wmma(a, b0, c[mi][0]);
            c[mi][1] = wmma(a, b1, c[mi][1]);
        }
    }

    _Float16* dst = (z == 0) ? Qo : (z == 1) ? Ko : Vt;
#pragma unroll
    for (int mi = 0; mi < 4; ++mi)
#pragma unroll
        for (int ni = 0; ni < 2; ++ni)
#pragma unroll
            for (int i = 0; i < 8; ++i) {
                int grow = m0 + mw + mi * 16 + (lane >> 4) * 8 + i;
                int gcol = n0 + nw + ni * 16 + (lane & 15);
                int b = grow >> 11, n = grow & 2047;   // NSEQ = 2048
                int h = gcol >> 6,  d = gcol & 63;     // DHEAD = 64
                _Float16 v = (_Float16)c[mi][ni][i];
                if (z == 2)
                    dst[((size_t)(b * NHEAD + h) * DHEAD + d) * NSEQ + n] = v;
                else
                    dst[((size_t)(b * NHEAD + h) * NSEQ + n) * DHEAD + d] = v;
            }
}

// ---------------------------------------------------------------------------
// Kernel 3: flash attention. Grid (16 q-tiles, 48 b*h). 8 waves x 16 queries.
// 16 WMMAs per wave per 64-key step (8 for S=QK^T, 8 for O+=PV).
// ---------------------------------------------------------------------------
__global__ __launch_bounds__(256) void attn(
        const _Float16* __restrict__ Q, const _Float16* __restrict__ K,
        const _Float16* __restrict__ Vt, _Float16* __restrict__ H) {
    __shared__ _Float16 Ks[64 * 64];        // [key][d]   -> B operand of Q@K^T
    __shared__ _Float16 Vs[64 * 64];        // [d][key]   -> B operand of P@V
    __shared__ _Float16 Ps[8 * 16 * 64];    // per-wave P scratch (C->A relayout)

    const int bh   = blockIdx.y;
    const int m0   = blockIdx.x * 128;
    const int tid  = threadIdx.x;
    const int wave = tid >> 5, lane = tid & 31;
    const _Float16* qp = Q  + (size_t)bh * NSEQ * DHEAD;
    const _Float16* kp = K  + (size_t)bh * NSEQ * DHEAD;
    const _Float16* vp = Vt + (size_t)bh * DHEAD * NSEQ;
    const int qrow = m0 + wave * 16;

    // Q fragments held in registers for the whole KV sweep (scale pre-folded).
    v16h qa0 = load_frag(qp,      qrow, DHEAD);   // d 0..31
    v16h qa1 = load_frag(qp + 32, qrow, DHEAD);   // d 32..63

    v8f o[4] = {};
    float mrun[8], lrun[8];
#pragma unroll
    for (int i = 0; i < 8; ++i) { mrun[i] = -1e30f; lrun[i] = 0.f; }
    _Float16* psw = Ps + wave * 16 * 64;

    for (int kv0 = 0; kv0 < NSEQ; kv0 += 64) {
        __syncthreads();
#pragma unroll
        for (int i = 0; i < 2; ++i) {       // stage 64x64 K and V^T tiles
            int idx = tid + i * 256;
            int row = idx >> 3, seg = (idx & 7) * 8;
            copy16_to_lds(&Ks[row * 64 + seg],
                          kp + (size_t)(kv0 + row) * DHEAD + seg);
            copy16_to_lds(&Vs[row * 64 + seg],
                          vp + (size_t)row * NSEQ + kv0 + seg);
        }
        if (kv0 + 64 < NSEQ) {  // prefetch next KV tile
            __builtin_prefetch(kp + (size_t)(kv0 + 64 + (tid >> 3)) * DHEAD + (tid & 7) * 8, 0, 3);
            __builtin_prefetch(vp + (size_t)(tid >> 3) * NSEQ + kv0 + 64 + (tid & 7) * 8, 0, 3);
        }
        async_fence();
        __syncthreads();

        // S = Q @ K^T : 16 x 64, f32
        v8f s[4];
#pragma unroll
        for (int ni = 0; ni < 4; ++ni) {
            v8f zc = {};
            zc    = wmma(qa0, load_frag(Ks,      16 * ni, 64), zc);
            s[ni] = wmma(qa1, load_frag(Ks + 32, 16 * ni, 64), zc);
        }

        // Online softmax per C-layout row slot i (row = 8*(lane>>4)+i).
#pragma unroll
        for (int i = 0; i < 8; ++i) {
            float rm = s[0][i];
#pragma unroll
            for (int ni = 1; ni < 4; ++ni) rm = fmaxf(rm, s[ni][i]);
#pragma unroll
            for (int off = 1; off < 16; off <<= 1)
                rm = fmaxf(rm, __shfl_xor(rm, off, 32));
            float mnew = fmaxf(mrun[i], rm);
            float corr = __expf(mrun[i] - mnew);
            mrun[i] = mnew;
            float rs = 0.f;
#pragma unroll
            for (int ni = 0; ni < 4; ++ni) {
                float p = __expf(s[ni][i] - mnew);
                s[ni][i] = p; rs += p;
            }
#pragma unroll
            for (int off = 1; off < 16; off <<= 1)
                rs += __shfl_xor(rs, off, 32);
            lrun[i] = lrun[i] * corr + rs;
#pragma unroll
            for (int di = 0; di < 4; ++di) o[di][i] *= corr;
            // spill P (C layout) to per-wave LDS in A layout order [row][key]
            int r = (lane >> 4) * 8 + i;
#pragma unroll
            for (int ni = 0; ni < 4; ++ni)
                psw[r * 64 + ni * 16 + (lane & 15)] = (_Float16)s[ni][i];
        }

        // O += P @ V : A frags from LDS, B frags from V^T tile
        v16h pa0 = load_frag(psw,      0, 64);    // keys 0..31
        v16h pa1 = load_frag(psw + 32, 0, 64);    // keys 32..63
#pragma unroll
        for (int di = 0; di < 4; ++di) {
            o[di] = wmma(pa0, load_frag(Vs,      16 * di, 64), o[di]);
            o[di] = wmma(pa1, load_frag(Vs + 32, 16 * di, 64), o[di]);
        }
    }

    const int b = bh / NHEAD, h = bh % NHEAD;
#pragma unroll
    for (int i = 0; i < 8; ++i) {
        float inv = 1.f / lrun[i];
        int row = qrow + (lane >> 4) * 8 + i;
#pragma unroll
        for (int di = 0; di < 4; ++di) {
            int col = h * DHEAD + di * 16 + (lane & 15);
            H[(size_t)(b * NSEQ + row) * DIMF + col] = (_Float16)(o[di][i] * inv);
        }
    }
}

// ---------------------------------------------------------------------------
// Kernel 4: output projection  out = H @ w_o + b_o  (f32 out).
// Workgroup tile 256x64, wave tile 64x32, all-f16 tiles staged async.
// ---------------------------------------------------------------------------
__global__ __launch_bounds__(256) void out_gemm(
        const _Float16* __restrict__ Hs, const _Float16* __restrict__ woT,
        const float* __restrict__ bias, float* __restrict__ out) {
    __shared__ _Float16 As[256 * 32];   // 16 KB
    __shared__ _Float16 Bs[64 * 32];    //  4 KB

    const int n0   = blockIdx.x * 64;
    const int m0   = blockIdx.y * 256;
    const int tid  = threadIdx.x;
    const int wave = tid >> 5, lane = tid & 31;
    const int mw   = (wave >> 1) * 64;
    const int nw   = (wave & 1) * 32;

    v8f c[4][2] = {};
    for (int k0 = 0; k0 < DIMF; k0 += 32) {
        __syncthreads();
#pragma unroll
        for (int i = 0; i < 4; ++i) {       // A: 256x32 halves, 4 b128/thread
            int idx = tid + i * 256;
            int row = idx >> 2, seg = (idx & 3) * 8;
            copy16_to_lds(&As[row * 32 + seg],
                          Hs + (size_t)(m0 + row) * DIMF + k0 + seg);
        }
        {
            int row = tid >> 2, seg = (tid & 3) * 8;
            copy16_to_lds(&Bs[row * 32 + seg],
                          woT + (size_t)(n0 + row) * DIMF + k0 + seg);
        }
        if (k0 + 32 < DIMF) {
            __builtin_prefetch(Hs + (size_t)(m0 + (tid >> 2)) * DIMF + k0 + 32 + (tid & 3) * 8, 0, 3);
            __builtin_prefetch(woT + (size_t)(n0 + (tid >> 2)) * DIMF + k0 + 32 + (tid & 3) * 8, 0, 3);
        }
        async_fence();
        __syncthreads();
        v16h b0 = load_frag(Bs, nw +  0, 32);
        v16h b1 = load_frag(Bs, nw + 16, 32);
#pragma unroll
        for (int mi = 0; mi < 4; ++mi) {
            v16h a = load_frag(As, mw + mi * 16, 32);
            c[mi][0] = wmma(a, b0, c[mi][0]);
            c[mi][1] = wmma(a, b1, c[mi][1]);
        }
    }
#pragma unroll
    for (int mi = 0; mi < 4; ++mi)
#pragma unroll
        for (int ni = 0; ni < 2; ++ni)
#pragma unroll
            for (int i = 0; i < 8; ++i) {
                int grow = m0 + mw + mi * 16 + (lane >> 4) * 8 + i;
                int gcol = n0 + nw + ni * 16 + (lane & 15);
                out[(size_t)grow * DIMF + gcol] = c[mi][ni][i] + bias[gcol];
            }
}

// ---------------------------------------------------------------------------
extern "C" void kernel_launch(void* const* d_in, const int* in_sizes, int n_in,
                              void* d_out, int out_size, void* d_ws, size_t ws_size,
                              hipStream_t stream) {
    (void)in_sizes; (void)n_in; (void)out_size; (void)ws_size;
    const float* x  = (const float*)d_in[0];
    const float* wq = (const float*)d_in[1];
    const float* wk = (const float*)d_in[2];
    const float* wv = (const float*)d_in[3];
    const float* wo = (const float*)d_in[4];
    const float* bo = (const float*)d_in[5];
    float* out = (float*)d_out;

    // Workspace layout (f16 elements): 4 transposed weights, x16, Q, K, V^T, H.
    // Total = 4*589824 + 5*6291456 = 33,816,576 halves = ~64.5 MB.
    _Float16* ws = (_Float16*)d_ws;
    const size_t WSZ = (size_t)DIMF * DIMF;                      // 589824
    const size_t ACT = (size_t)MROWS * DIMF;                     // 6291456
    _Float16* wT  = ws;
    _Float16* X16 = ws + 4 * WSZ;
    _Float16* Qb  = X16 + ACT;
    _Float16* Kb  = Qb + ACT;   // per-head layouts are same total size
    _Float16* Vb  = Kb + ACT;
    _Float16* Hb  = Vb + ACT;

    convert_weights<<<dim3(WSZ / 256, 4), 256, 0, stream>>>(wq, wk, wv, wo, wT);
    convert_x<<<dim3(ACT / (256 * 8)), 256, 0, stream>>>(x, X16);
    qkv_gemm<<<dim3(DIMF / 64, MROWS / 256, 3), 256, 0, stream>>>(X16, wT, Qb, Kb, Vb);
    attn<<<dim3(NSEQ / 128, BATCH * NHEAD), 256, 0, stream>>>(Qb, Kb, Vb, Hb);
    out_gemm<<<dim3(DIMF / 64, MROWS / 256), 256, 0, stream>>>(Hb, wT + 3 * WSZ, bo, out);
}
